// LSTM_76819785056938
// MI455X (gfx1250) — matrix operand order repaired and checked by reading, not compiled, not run on hardware
//
#include <hip/hip_runtime.h>
#include <hip/hip_bf16.h>

// ---------------------------------------------------------------------------
// MI455X (gfx1250) fused LSTM policy network.
//  - bf16 WMMA (16x16x32) for every GEMM including the recurrent h @ W_hh^T
//  - W_hh (8MB bf16) stays resident in the 192MB L2 across the 512 steps
//  - x-gates precomputed for all T in one big WMMA GEMM (fp32 accumulate)
//  - per-timestep kernel: 16 blocks x 8 waves, LDS exchange of i/f/g/o gates
// ---------------------------------------------------------------------------

typedef __bf16 bf16_t;
typedef __attribute__((ext_vector_type(16))) __bf16 v16bf;
typedef __attribute__((ext_vector_type(8)))  __bf16 v8bf;
typedef __attribute__((ext_vector_type(8)))  float  v8f;

union FragBF { v16bf v; v8bf h[2]; };

// A-fragment (16x32, M x K), row-major source with leading dim ld.
// Layout used here: lane l -> row M = l%16; k-half kh = l/16;
//   VGPR0-7 = K[16*kh .. 16*kh+15] (contiguous, matching B)
__device__ __forceinline__ v16bf load_a_frag(const bf16_t* __restrict__ base, int ld) {
    const int lane = threadIdx.x & 31;
    const int row  = lane & 15;
    const int kh   = lane >> 4;
    const bf16_t* p = base + (size_t)row * ld + kh * 16;
    FragBF f;
    f.h[0] = *(const v8bf*)(p);
    f.h[1] = *(const v8bf*)(p + 8);
    return f.v;
}

// B-fragment (32x16, K x N) where B = W^T and W is [N,K] row-major.
// Layout used here: lane l -> col N = l%16; lanes 0-15 hold K=0..15, 16-31 hold 16..31,
// i.e. per lane one contiguous run of 16 K-values: W[n][16*kh .. 16*kh+15].
__device__ __forceinline__ v16bf load_b_frag(const bf16_t* __restrict__ base, int ld) {
    const int lane = threadIdx.x & 31;
    const bf16_t* p = base + (size_t)(lane & 15) * ld + (lane >> 4) * 16;
    FragBF f;
    f.h[0] = *(const v8bf*)(p);
    f.h[1] = *(const v8bf*)(p + 8);
    return f.v;
}

__device__ __forceinline__ float fsigmoid(float x) { return 1.0f / (1.0f + __expf(-x)); }

// ---------------------------------------------------------------------------
// Generic WMMA GEMM:  Out[M,N] = act( A[M,K] @ W[N,K]^T + bias0 + bias1 )
// Block = 256 threads (8 waves), tile 128(M) x 64(N), K step 32.
// Wave w: M-tiles {2*(w%4), 2*(w%4)+1}, N-tiles {2*(w/4), 2*(w/4)+1}.
// ---------------------------------------------------------------------------
template<bool RELU, bool OUT_BF16>
__global__ __launch_bounds__(256)
void gemm_wmma(const bf16_t* __restrict__ A, const bf16_t* __restrict__ W,
               const float* __restrict__ bias0, const float* __restrict__ bias1,
               void* __restrict__ Out, int M, int N, int K)
{
    const int lane  = threadIdx.x & 31;
    const int wave  = threadIdx.x >> 5;
    const int mBase = blockIdx.y * 128 + (wave & 3) * 32;
    const int nBase = blockIdx.x * 64  + (wave >> 2) * 32;

    v8f acc[2][2] = {};

    for (int k = 0; k < K; k += 32) {
        if (k + 32 < K) {   // speculative prefetch of next K slice (global_prefetch_b8)
            __builtin_prefetch(A + (size_t)mBase * K + k + 32, 0, 0);
            __builtin_prefetch(W + (size_t)nBase * K + k + 32, 0, 0);
        }
        v16bf a0 = load_a_frag(A + (size_t)(mBase     ) * K + k, K);
        v16bf a1 = load_a_frag(A + (size_t)(mBase + 16) * K + k, K);
        v16bf b0 = load_b_frag(W + (size_t)(nBase     ) * K + k, K);
        v16bf b1 = load_b_frag(W + (size_t)(nBase + 16) * K + k, K);
        acc[0][0] = __builtin_amdgcn_wmma_f32_16x16x32_bf16(false, a0, false, b0, (short)0, acc[0][0], false, false);
        acc[0][1] = __builtin_amdgcn_wmma_f32_16x16x32_bf16(false, a0, false, b1, (short)0, acc[0][1], false, false);
        acc[1][0] = __builtin_amdgcn_wmma_f32_16x16x32_bf16(false, a1, false, b0, (short)0, acc[1][0], false, false);
        acc[1][1] = __builtin_amdgcn_wmma_f32_16x16x32_bf16(false, a1, false, b1, (short)0, acc[1][1], false, false);
    }

    // Epilogue.  C/D layout: lane l -> N = l%16, M = r + 8*(l/16) for VGPR r.
    const int mLane = 8 * (lane >> 4);
    const int nCol  = lane & 15;
    #pragma unroll
    for (int mi = 0; mi < 2; ++mi) {
        #pragma unroll
        for (int ni = 0; ni < 2; ++ni) {
            const int n = nBase + ni * 16 + nCol;
            float bsum = 0.0f;
            if (bias0) bsum += bias0[n];
            if (bias1) bsum += bias1[n];
            #pragma unroll
            for (int r = 0; r < 8; ++r) {
                const int m = mBase + mi * 16 + mLane + r;
                float v = acc[mi][ni][r] + bsum;
                if (RELU) v = fmaxf(v, 0.0f);
                if (OUT_BF16) ((bf16_t*)Out)[(size_t)m * N + n] = (bf16_t)v;
                else          ((float*) Out)[(size_t)m * N + n] = v;
            }
        }
    }
}

// ---------------------------------------------------------------------------
// One LSTM time step.  grid = NH/64 = 16 blocks; block handles 64 h-columns
// (=> 4*64 = 256 gate columns so i/f/g/o for those h-columns live in-block).
// Wave w: gate g = w/2, N-tile pair (w%2)*2.., both M-tiles (B=32).
// ---------------------------------------------------------------------------
__global__ __launch_bounds__(256)
void lstm_step(const bf16_t* __restrict__ hprev,   // [32,1024] bf16
               bf16_t*       __restrict__ hnext,   // [32,1024] bf16
               float*        __restrict__ c,       // [32,1024] f32 (in/out)
               const float*  __restrict__ xg,      // [B*T,4096] f32 (biases folded)
               const bf16_t* __restrict__ Whh,     // [4096,1024] bf16 row-major
               bf16_t*       __restrict__ hs,      // [B*T,1024] bf16
               int t)
{
    constexpr int T  = 512;
    constexpr int NH = 1024;
    __shared__ float gbuf[4 * 32 * 64];            // 32 KB: gates[g][b][localcol]

    const int lane = threadIdx.x & 31;
    const int wave = threadIdx.x >> 5;
    const int j0   = blockIdx.x * 64;              // h-column base
    const int g    = wave >> 1;                    // gate 0..3 (i,f,g,o)
    const int np   = (wave & 1) * 2;               // N-tile pair within 64 cols

    v8f acc[2][2] = {};
    const size_t rowB0 = (size_t)(g * NH + j0 + (np    ) * 16) * NH;
    const size_t rowB1 = (size_t)(g * NH + j0 + (np + 1) * 16) * NH;

    for (int k = 0; k < NH; k += 32) {
        v16bf a0 = load_a_frag(hprev + k, NH);                 // batch rows 0..15
        v16bf a1 = load_a_frag(hprev + (size_t)16 * NH + k, NH); // rows 16..31
        v16bf b0 = load_b_frag(Whh + rowB0 + k, NH);
        v16bf b1 = load_b_frag(Whh + rowB1 + k, NH);
        acc[0][0] = __builtin_amdgcn_wmma_f32_16x16x32_bf16(false, a0, false, b0, (short)0, acc[0][0], false, false);
        acc[0][1] = __builtin_amdgcn_wmma_f32_16x16x32_bf16(false, a0, false, b1, (short)0, acc[0][1], false, false);
        acc[1][0] = __builtin_amdgcn_wmma_f32_16x16x32_bf16(false, a1, false, b0, (short)0, acc[1][0], false, false);
        acc[1][1] = __builtin_amdgcn_wmma_f32_16x16x32_bf16(false, a1, false, b1, (short)0, acc[1][1], false, false);
    }

    // Scatter gate pre-activations (+ precomputed x-gates) into LDS.
    const int mLane = 8 * (lane >> 4);
    const int nCol  = lane & 15;
    #pragma unroll
    for (int mi = 0; mi < 2; ++mi) {
        #pragma unroll
        for (int ni = 0; ni < 2; ++ni) {
            const int lj = (np + ni) * 16 + nCol;          // 0..63
            #pragma unroll
            for (int r = 0; r < 8; ++r) {
                const int b = mi * 16 + mLane + r;         // batch 0..31
                const float xv = xg[((size_t)b * T + t) * 4096 + (size_t)g * NH + j0 + lj];
                gbuf[g * 2048 + b * 64 + lj] = acc[mi][ni][r] + xv;
            }
        }
    }
    __syncthreads();

    // Elementwise cell update: 2048 (b, h-col) pairs across 256 threads.
    for (int e = threadIdx.x; e < 2048; e += 256) {
        const int b = e >> 6;
        const int j = e & 63;
        const float iv = gbuf[0 * 2048 + e];
        const float fv = gbuf[1 * 2048 + e];
        const float gv = gbuf[2 * 2048 + e];
        const float ov = gbuf[3 * 2048 + e];
        const size_t cidx = (size_t)b * NH + j0 + j;
        const float cn = fsigmoid(fv) * c[cidx] + fsigmoid(iv) * tanhf(gv);
        const float hn = fsigmoid(ov) * tanhf(cn);
        c[cidx] = cn;
        hnext[cidx] = (bf16_t)hn;
        hs[((size_t)b * T + t) * (size_t)NH + j0 + j] = (bf16_t)hn;
    }
}

// ---------------------------------------------------------------------------
// Softmax over 64 logits per row; one wave32 per row (2 elems/lane).
// ---------------------------------------------------------------------------
__global__ __launch_bounds__(256)
void softmax64(const float* __restrict__ logits, float* __restrict__ probs, int rows)
{
    const int wave = threadIdx.x >> 5;
    const int lane = threadIdx.x & 31;
    const int row  = blockIdx.x * 8 + wave;
    if (row >= rows) return;
    const float* p = logits + (size_t)row * 64;
    const float x0 = p[lane], x1 = p[lane + 32];
    float mx = fmaxf(x0, x1);
    for (int off = 16; off; off >>= 1) mx = fmaxf(mx, __shfl_xor(mx, off, 32));
    const float e0 = __expf(x0 - mx), e1 = __expf(x1 - mx);
    float s = e0 + e1;
    for (int off = 16; off; off >>= 1) s += __shfl_xor(s, off, 32);
    const float inv = 1.0f / s;
    float* o = probs + (size_t)row * 64;
    o[lane]      = e0 * inv;
    o[lane + 32] = e1 * inv;
}

// ---------------------------------------------------------------------------
// Value head GEMV: out[row] = dot(V[row,0..511], w) + b ; one wave per row.
// ---------------------------------------------------------------------------
__global__ __launch_bounds__(256)
void gemv_val(const bf16_t* __restrict__ V, const bf16_t* __restrict__ w,
              const float* __restrict__ b, float* __restrict__ out, int rows)
{
    const int wave = threadIdx.x >> 5;
    const int lane = threadIdx.x & 31;
    const int row  = blockIdx.x * 8 + wave;
    if (row >= rows) return;
    const bf16_t* p = V + (size_t)row * 512;
    float s = 0.0f;
    #pragma unroll
    for (int i = 0; i < 16; ++i) {
        const int k = lane + i * 32;
        s += (float)p[k] * (float)w[k];
    }
    for (int off = 16; off; off >>= 1) s += __shfl_xor(s, off, 32);
    if (lane == 0) out[row] = s + b[0];
}

// ---------------------------------------------------------------------------
// fp32 -> bf16 conversion (grid-stride).
// ---------------------------------------------------------------------------
__global__ __launch_bounds__(256)
void f32_to_bf16(const float* __restrict__ src, bf16_t* __restrict__ dst, size_t n)
{
    size_t i = (size_t)blockIdx.x * blockDim.x + threadIdx.x;
    const size_t stride = (size_t)gridDim.x * blockDim.x;
    for (; i < n; i += stride) dst[i] = (bf16_t)src[i];
}

// ---------------------------------------------------------------------------

extern "C" void kernel_launch(void* const* d_in, const int* in_sizes, int n_in,
                              void* d_out, int out_size, void* d_ws, size_t ws_size,
                              hipStream_t stream)
{
    (void)in_sizes; (void)n_in; (void)out_size; (void)ws_size;
    constexpr int B = 32, T = 512, IN = 256, PRE = 512, LH = 512, NH = 1024,
                  ACT_H = 512, NPOL = 64;
    constexpr int BT = B * T;

    const float* x        = (const float*)d_in[0];
    const float* pre_W0   = (const float*)d_in[1];
    const float* pre_b0   = (const float*)d_in[2];
    const float* pre_W1   = (const float*)d_in[3];
    const float* pre_b1   = (const float*)d_in[4];
    const float* w_ih     = (const float*)d_in[5];
    const float* w_hh     = (const float*)d_in[6];
    const float* bias_ih  = (const float*)d_in[7];
    const float* bias_hh  = (const float*)d_in[8];
    const float* act_W0   = (const float*)d_in[9];
    const float* act_b0   = (const float*)d_in[10];
    const float* act_W1   = (const float*)d_in[11];
    const float* act_b1   = (const float*)d_in[12];
    const float* val_W0   = (const float*)d_in[13];
    const float* val_b0   = (const float*)d_in[14];
    const float* val_W1   = (const float*)d_in[15];
    const float* val_b1   = (const float*)d_in[16];

    // ---- workspace carve-up (256B aligned) ----
    char* wsp = (char*)d_ws;
    auto carve = [&](size_t bytes) { char* p = wsp; wsp += (bytes + 255) & ~(size_t)255; return p; };
    bf16_t* x_bf    = (bf16_t*)carve((size_t)BT * IN * 2);
    bf16_t* pW0_bf  = (bf16_t*)carve((size_t)PRE * IN * 2);
    bf16_t* pW1_bf  = (bf16_t*)carve((size_t)LH * PRE * 2);
    bf16_t* Wih_bf  = (bf16_t*)carve((size_t)4 * NH * LH * 2);
    bf16_t* Whh_bf  = (bf16_t*)carve((size_t)4 * NH * NH * 2);
    bf16_t* aW0_bf  = (bf16_t*)carve((size_t)ACT_H * NH * 2);
    bf16_t* aW1_bf  = (bf16_t*)carve((size_t)NPOL * ACT_H * 2);
    bf16_t* vW0_bf  = (bf16_t*)carve((size_t)ACT_H * NH * 2);
    bf16_t* vW1_bf  = (bf16_t*)carve((size_t)ACT_H * 2);
    bf16_t* hpre    = (bf16_t*)carve((size_t)BT * PRE * 2);
    bf16_t* u_bf    = (bf16_t*)carve((size_t)BT * LH * 2);
    float*  xg      = (float*) carve((size_t)BT * 4 * NH * 4);
    bf16_t* hbuf    = (bf16_t*)carve((size_t)2 * B * NH * 2);
    float*  cstate  = (float*) carve((size_t)B * NH * 4);
    bf16_t* hs      = (bf16_t*)carve((size_t)BT * NH * 2);
    bf16_t* a_bf    = (bf16_t*)carve((size_t)BT * ACT_H * 2);
    bf16_t* v_bf    = (bf16_t*)carve((size_t)BT * ACT_H * 2);
    float*  logits  = (float*) carve((size_t)BT * NPOL * 4);

    float* probs_out = (float*)d_out;                 // [BT, 64]
    float* value_out = (float*)d_out + (size_t)BT * NPOL;  // [BT]

    // ---- fp32 -> bf16 conversions ----
    auto cvt = [&](const float* s, bf16_t* d, size_t n) {
        const int blocks = (int)((n + 256 * 8 - 1) / (256 * 8));
        f32_to_bf16<<<blocks, 256, 0, stream>>>(s, d, n);
    };
    cvt(x,      x_bf,   (size_t)BT * IN);
    cvt(pre_W0, pW0_bf, (size_t)PRE * IN);
    cvt(pre_W1, pW1_bf, (size_t)LH * PRE);
    cvt(w_ih,   Wih_bf, (size_t)4 * NH * LH);
    cvt(w_hh,   Whh_bf, (size_t)4 * NH * NH);
    cvt(act_W0, aW0_bf, (size_t)ACT_H * NH);
    cvt(act_W1, aW1_bf, (size_t)NPOL * ACT_H);
    cvt(val_W0, vW0_bf, (size_t)ACT_H * NH);
    cvt(val_W1, vW1_bf, (size_t)ACT_H);

    // ---- zero initial LSTM state (bf16 zero == 0x0000, f32 zero == 0x0) ----
    hipMemsetAsync(hbuf,   0, (size_t)2 * B * NH * 2, stream);
    hipMemsetAsync(cstate, 0, (size_t)B * NH * 4, stream);

    // ---- pre-DNN + input-gate GEMMs (WMMA bf16, f32 accumulate) ----
    dim3 blk(256);
    // h = relu(x @ pre_W0^T + pre_b0)
    gemm_wmma<true,  true ><<<dim3(PRE / 64,  BT / 128), blk, 0, stream>>>(
        x_bf, pW0_bf, pre_b0, nullptr, hpre, BT, PRE, IN);
    // u = h @ pre_W1^T + pre_b1
    gemm_wmma<false, true ><<<dim3(LH / 64,   BT / 128), blk, 0, stream>>>(
        hpre, pW1_bf, pre_b1, nullptr, u_bf, BT, LH, PRE);
    // x_gates = u @ W_ih^T + (bias_ih + bias_hh)   (fp32 for the recurrence)
    gemm_wmma<false, false><<<dim3(4 * NH / 64, BT / 128), blk, 0, stream>>>(
        u_bf, Wih_bf, bias_ih, bias_hh, xg, BT, 4 * NH, LH);

    // ---- sequential LSTM: 512 dependent steps; W_hh lives in L2 ----
    for (int t = 0; t < T; ++t) {
        bf16_t* hp = hbuf + (size_t)(t & 1) * B * NH;
        bf16_t* hn = hbuf + (size_t)((t + 1) & 1) * B * NH;
        lstm_step<<<dim3(NH / 64), blk, 0, stream>>>(hp, hn, cstate, xg, Whh_bf, hs, t);
    }

    // ---- policy head ----
    gemm_wmma<true,  true ><<<dim3(ACT_H / 64, BT / 128), blk, 0, stream>>>(
        hs, aW0_bf, act_b0, nullptr, a_bf, BT, ACT_H, NH);
    gemm_wmma<false, false><<<dim3(NPOL / 64,  BT / 128), blk, 0, stream>>>(
        a_bf, aW1_bf, act_b1, nullptr, logits, BT, NPOL, ACT_H);
    softmax64<<<dim3(BT / 8), blk, 0, stream>>>(logits, probs_out, BT);

    // ---- value head ----
    gemm_wmma<true,  true ><<<dim3(ACT_H / 64, BT / 128), blk, 0, stream>>>(
        hs, vW0_bf, val_b0, nullptr, v_bf, BT, ACT_H, NH);
    gemv_val<<<dim3(BT / 8), blk, 0, stream>>>(v_bf, vW1_bf, val_b1, value_out, BT);
}
